// MLP_17961553232283
// MI455X (gfx1250) — compile-verified, matrix operand order checked by reading, not run on hardware
//
#include <hip/hip_runtime.h>
#include <hip/hip_bf16.h>

// ---------------- problem constants ----------------
#define T_TOK   8192          // tokens = 2048*4
#define HID     1024          // hidden
#define FFN     2048          // GLU width; w1 projects to 2*FFN
#define W1COLS  (2*FFN)       // 4096
#define NEXP    8
#define TILE_M  32            // tokens per expert block
#define FCHUNK  128           // FFN chunk per fused iteration

typedef __attribute__((ext_vector_type(16))) __bf16 v16bf;
typedef __attribute__((ext_vector_type(8)))  float  v8f;
typedef int v4i __attribute__((vector_size(16)));   // matches builtin param type

// ---- CDNA5 async-to-LDS path (guarded; falls back to sync LDS stores) ----
#if defined(__has_builtin)
#if __has_builtin(__builtin_amdgcn_global_load_async_to_lds_b128) && \
    __has_builtin(__builtin_amdgcn_s_wait_asynccnt)
#define USE_ASYNC_LDS 1
#endif
#endif
#ifndef USE_ASYNC_LDS
#define USE_ASYNC_LDS 0
#endif

// copy 16 bytes global -> LDS (per lane)
__device__ __forceinline__ void cp16(unsigned short* ldst,
                                     const unsigned short* gsrc) {
#if USE_ASYNC_LDS
  __builtin_amdgcn_global_load_async_to_lds_b128(
      (__attribute__((address_space(1))) v4i*)gsrc,
      (__attribute__((address_space(3))) v4i*)ldst, 0, 0);
#else
  *(uint4*)ldst = *(const uint4*)gsrc;
#endif
}

__device__ __forceinline__ void stage_wait() {
#if USE_ASYNC_LDS
  __builtin_amdgcn_s_wait_asynccnt(0);
#endif
  __syncthreads();
}

// fp32 -> bf16 round-to-nearest-even (bit trick)
__device__ __forceinline__ unsigned short f2bf(float f) {
  unsigned int u = __float_as_uint(f);
  u += 0x7FFFu + ((u >> 16) & 1u);
  return (unsigned short)(u >> 16);
}

__device__ __forceinline__ v8f zero8() {
  v8f z = {0.f, 0.f, 0.f, 0.f, 0.f, 0.f, 0.f, 0.f};
  return z;
}

// Load a 16x16x32 bf16 WMMA A/B fragment from LDS.
// ISA layout: lane half hl = lane>>4; element i holds K = 16*(i/8) + 8*hl + (i%8)
// => two contiguous 16B runs at +8*hl and +16+8*hl elements from the row base.
__device__ __forceinline__ v16bf load_frag16(const unsigned short* p, int hl) {
  union { v16bf v; uint4 q[2]; } u;
  u.q[0] = *(const uint4*)(p + 8 * hl);
  u.q[1] = *(const uint4*)(p + 16 + 8 * hl);
  return u.v;
}

__device__ __forceinline__ v8f wmma_bf16(v16bf a, v16bf b, v8f c) {
  return __builtin_amdgcn_wmma_f32_16x16x32_bf16(false, a, false, b, (short)0, c,
                                                 false, false);
}

// ---------------- kernel 0: zero output + counters ----------------
__global__ __launch_bounds__(256) void moe_init_kernel(float4* __restrict__ out4,
                                                       int n4,
                                                       int* __restrict__ counts) {
  int i = blockIdx.x * blockDim.x + threadIdx.x;
  if (blockIdx.x == 0 && threadIdx.x < NEXP) counts[threadIdx.x] = 0;
  float4 z; z.x = 0.f; z.y = 0.f; z.z = 0.f; z.w = 0.f;
  for (int idx = i; idx < n4; idx += gridDim.x * blockDim.x) out4[idx] = z;
}

// ---------------- weight pre-swizzle: fp32 [k][col] -> bf16 [ktile][col][32k] ----
// w1s[e][kt(32)][col(4096)][kr(32)]  (HID = 32 ktiles * 32)
__global__ __launch_bounds__(256) void conv_w1_kernel(
    const float* __restrict__ w1, unsigned short* __restrict__ w1s) {
  int gid = blockIdx.x * 256 + threadIdx.x;  // 8*32*4096 threads exactly
  int col = gid & (W1COLS - 1);
  int ekt = gid >> 12;                       // e*32 + kt
  const float* src = w1 + ((size_t)ekt * 32) * W1COLS + col;
  unsigned short* dst = w1s + ((size_t)ekt * W1COLS + col) * 32;
  unsigned int tmp[16];
#pragma unroll
  for (int kr = 0; kr < 32; kr += 2) {
    unsigned int lo = f2bf(src[(size_t)kr * W1COLS]);
    unsigned int hi = f2bf(src[(size_t)(kr + 1) * W1COLS]);
    tmp[kr >> 1] = lo | (hi << 16);
  }
#pragma unroll
  for (int j = 0; j < 4; j++) ((uint4*)dst)[j] = *(uint4*)&tmp[j * 4];
}

// w2s[e][kt2(64)][col(1024)][kr(32)]  (FFN = 64 ktiles * 32)
__global__ __launch_bounds__(256) void conv_w2_kernel(
    const float* __restrict__ w2, unsigned short* __restrict__ w2s) {
  int gid = blockIdx.x * 256 + threadIdx.x;  // 8*64*1024 threads exactly
  int col = gid & (HID - 1);
  int ekt = gid >> 10;                       // e*64 + kt2
  const float* src = w2 + ((size_t)ekt * 32) * HID + col;
  unsigned short* dst = w2s + ((size_t)ekt * HID + col) * 32;
  unsigned int tmp[16];
#pragma unroll
  for (int kr = 0; kr < 32; kr += 2) {
    unsigned int lo = f2bf(src[(size_t)kr * HID]);
    unsigned int hi = f2bf(src[(size_t)(kr + 1) * HID]);
    tmp[kr >> 1] = lo | (hi << 16);
  }
#pragma unroll
  for (int j = 0; j < 4; j++) ((uint4*)dst)[j] = *(uint4*)&tmp[j * 4];
}

// ---------------- kernel 1: router + x->bf16 ----------------
__global__ __launch_bounds__(256) void moe_router_kernel(
    const float* __restrict__ x, const float* __restrict__ wr,
    int* __restrict__ counts, int* __restrict__ tok_list,
    float* __restrict__ gate_list, unsigned short* __restrict__ xb) {
  __shared__ float wrs[HID * NEXP];  // 32KB
  const int tid = threadIdx.x;
  for (int i = tid; i < HID * NEXP; i += 256) wrs[i] = wr[i];
  __syncthreads();

  const int wid = tid >> 5, lane = tid & 31;
  const int t = blockIdx.x * 8 + wid;  // one token per wave

  float acc[NEXP];
#pragma unroll
  for (int e = 0; e < NEXP; e++) acc[e] = 0.f;

  const float* xrow = x + (size_t)t * HID;
  unsigned short* xbrow = xb + (size_t)t * HID;
  for (int h2 = lane * 2; h2 < HID; h2 += 64) {  // pair per lane, packed store
    float xv0 = xrow[h2], xv1 = xrow[h2 + 1];
    unsigned int pk = (unsigned int)f2bf(xv0) | ((unsigned int)f2bf(xv1) << 16);
    *(unsigned int*)(xbrow + h2) = pk;
#pragma unroll
    for (int e = 0; e < NEXP; e++)
      acc[e] += xv0 * wrs[h2 * NEXP + e] + xv1 * wrs[(h2 + 1) * NEXP + e];
  }
#pragma unroll
  for (int off = 16; off > 0; off >>= 1) {
#pragma unroll
    for (int e = 0; e < NEXP; e++) acc[e] += __shfl_xor(acc[e], off, 32);
  }

  if (lane == 0) {
    int e0 = 0; float v0 = acc[0];
#pragma unroll
    for (int e = 1; e < NEXP; e++) if (acc[e] > v0) { v0 = acc[e]; e0 = e; }
    int e1 = (e0 == 0) ? 1 : 0; float v1 = acc[e1];
#pragma unroll
    for (int e = 0; e < NEXP; e++)
      if (e != e0 && acc[e] > v1) { v1 = acc[e]; e1 = e; }
    float g0 = 1.f / (1.f + __expf(v1 - v0));  // softmax over top-2 (v0 >= v1)
    float g1 = 1.f - g0;
    int p0 = atomicAdd(&counts[e0], 1);
    tok_list[e0 * T_TOK + p0] = t; gate_list[e0 * T_TOK + p0] = g0;
    int p1 = atomicAdd(&counts[e1], 1);
    tok_list[e1 * T_TOK + p1] = t; gate_list[e1 * T_TOK + p1] = g1;
  }
}

// ---------------- kernel 2: fused expert (GEMM1 + SwiGLU + GEMM2) ----------------
union alignas(16) SmemU {
  struct {
    unsigned short xs[2][TILE_M][32];   // 4KB  double-buffered X tile
    unsigned short w1t[2][256][32];     // 32KB double-buffered w1 [vcol][kr]
  } g1;                                 // 36KB
  float abuf[TILE_M][256];              // 32KB a (0..127) / g (128..255) fp32
  unsigned short w2t[2][256][32];       // 32KB double-buffered w2 quarter
};

__device__ __forceinline__ void stage_g1(SmemU* u, int buf, int k, int tid,
                                         const unsigned short* __restrict__ xb,
                                         const int* tok_s,
                                         const unsigned short* __restrict__ w1s_e,
                                         int fc) {
  const int kt = k >> 5;
  const unsigned short* a_src = w1s_e + (((size_t)kt * W1COLS) + fc) * 32;
  const unsigned short* g_src = w1s_e + (((size_t)kt * W1COLS) + FFN + fc) * 32;
  unsigned short* a_dst = &u->g1.w1t[buf][0][0];
  unsigned short* g_dst = &u->g1.w1t[buf][128][0];
#pragma unroll
  for (int r = 0; r < 2; r++) {  // 2 * 256 chunks of 16B = 8KB each
    int i = (tid + r * 256) * 8;
    cp16(a_dst + i, a_src + i);
    cp16(g_dst + i, g_src + i);
  }
  if (tid < 128) {  // X tile: 32 rows * 64B (gathered rows)
    int row = tid >> 2, off = (tid & 3) * 8;
    cp16(&u->g1.xs[buf][row][off],
         xb + (size_t)tok_s[row] * HID + k + off);
  }
}

__device__ __forceinline__ void stage_g2(SmemU* u, int buf,
                                         const unsigned short* __restrict__ w2s_e,
                                         int kt2, int qq, int tid) {
  const unsigned short* src = w2s_e + (((size_t)kt2 * HID) + qq * 256) * 32;
  unsigned short* dst = &u->w2t[buf][0][0];
#pragma unroll
  for (int r = 0; r < 4; r++) {  // 1024 chunks of 16B = 16KB
    int i = (tid + r * 256) * 8;
    cp16(dst + i, src + i);
  }
}

__global__ __launch_bounds__(256) void moe_expert_kernel(
    const unsigned short* __restrict__ xb,
    const unsigned short* __restrict__ w1s,
    const unsigned short* __restrict__ w2s, const int* __restrict__ counts,
    const int* __restrict__ tok_list, const float* __restrict__ gate_list,
    float* __restrict__ out) {
  __shared__ SmemU u;
  __shared__ alignas(16) unsigned short actb[TILE_M][FCHUNK];  // 8KB
  __shared__ int tok_s[TILE_M];
  __shared__ float gate_s[TILE_M];

  const int e = blockIdx.x >> 8;  // 256 tile slots per expert
  const int tile = blockIdx.x & 255;
  const int cnt = counts[e];
  if (tile * TILE_M >= cnt) return;  // block-uniform exit: EXEC all-ones for WMMA

  const int tid = threadIdx.x;
  const int wid = tid >> 5;
  const int lane = tid & 31;
  const int mrow = lane & 15;
  const int hl = lane >> 4;

  if (tid < TILE_M) {
    int slot = tile * TILE_M + tid;
    if (slot < cnt) {
      tok_s[tid] = tok_list[e * T_TOK + slot];
      gate_s[tid] = gate_list[e * T_TOK + slot];
    } else {
      tok_s[tid] = 0;   // real data, gate 0 => contributes exactly 0
      gate_s[tid] = 0.f;
    }
  }
  __syncthreads();

  const unsigned short* w1s_e = w1s + (size_t)e * HID * W1COLS;  // [kt][col][kr]
  const unsigned short* w2s_e = w2s + (size_t)e * FFN * HID;     // [kt2][col][kr]

  // out accumulators: 32 tokens x 1024 cols; wave owns col (q>>1)*256+wid*32+(q&1)*16
  v8f acc[2][8];
#pragma unroll
  for (int mt = 0; mt < 2; mt++)
#pragma unroll
    for (int q = 0; q < 8; q++) acc[mt][q] = zero8();

  for (int fc = 0; fc < FFN; fc += FCHUNK) {
    // ===== GEMM1: [32 x 256 vcols], vcol<128 -> a(fc+vc), else g(FFN+fc+vc-128)
    v8f g1acc[2][2];
#pragma unroll
    for (int mt = 0; mt < 2; mt++)
#pragma unroll
      for (int nt = 0; nt < 2; nt++) g1acc[mt][nt] = zero8();

    stage_g1(&u, 0, 0, tid, xb, tok_s, w1s_e, fc);  // prologue
    for (int ki = 0; ki < HID / 32; ki++) {
      const int cur = ki & 1;
      stage_wait();  // buf[cur] ready; all waves done reading buf[cur^1]
      if (ki + 1 < HID / 32)
        stage_g1(&u, cur ^ 1, (ki + 1) * 32, tid, xb, tok_s, w1s_e, fc);
#pragma unroll
      for (int mt = 0; mt < 2; mt++) {
        v16bf af = load_frag16(&u.g1.xs[cur][mt * 16 + mrow][0], hl);
#pragma unroll
        for (int nt = 0; nt < 2; nt++) {
          v16bf bf = load_frag16(&u.g1.w1t[cur][wid * 32 + nt * 16 + mrow][0], hl);
          g1acc[mt][nt] = wmma_bf16(af, bf, g1acc[mt][nt]);
        }
      }
    }
    __syncthreads();  // all WMMA LDS reads done before aliasing with abuf

    // spill a,g to fp32 LDS (C layout: VGPR j holds M = j + 8*hl, N = mrow)
#pragma unroll
    for (int mt = 0; mt < 2; mt++)
#pragma unroll
      for (int nt = 0; nt < 2; nt++)
#pragma unroll
        for (int j = 0; j < 8; j++)
          u.abuf[mt * 16 + hl * 8 + j][wid * 32 + nt * 16 + mrow] =
              g1acc[mt][nt][j];
    __syncthreads();

    // SwiGLU: act = a * g * rcp(1+exp(-g)); 16 contiguous cols per thread
    {
      int base = tid * 16;              // 32*128 = 4096 elements
      int r = base >> 7, c = base & 127;
      unsigned int pk[8];
#pragma unroll
      for (int j = 0; j < 16; j += 2) {
        float a0 = u.abuf[r][c + j],     g0 = u.abuf[r][FCHUNK + c + j];
        float a1 = u.abuf[r][c + j + 1], g1v = u.abuf[r][FCHUNK + c + j + 1];
        float s0 = a0 * g0 * __builtin_amdgcn_rcpf(1.f + __expf(-g0));
        float s1 = a1 * g1v * __builtin_amdgcn_rcpf(1.f + __expf(-g1v));
        pk[j >> 1] = (unsigned int)f2bf(s0) | ((unsigned int)f2bf(s1) << 16);
      }
      uint4* dst = (uint4*)&actb[r][c];
      dst[0] = *(uint4*)&pk[0];
      dst[1] = *(uint4*)&pk[4];
    }
    __syncthreads();

    // ===== GEMM2 rank-128 update: acc += act[32x128] @ w2[fc..fc+128, 0..1024]
    // 16 stages: s = k2tile(4) * 4 + quarter(4); 16KB each, double-buffered
    stage_g2(&u, 0, w2s_e, (fc >> 5), 0, tid);  // prologue (k2=0, qq=0)
#pragma unroll 1
    for (int s = 0; s < 16; s++) {
      const int cur = s & 1;
      stage_wait();
      if (s + 1 < 16)
        stage_g2(&u, cur ^ 1, w2s_e, ((fc + ((s + 1) >> 2) * 32) >> 5),
                 (s + 1) & 3, tid);
      const int k2 = (s >> 2) * 32;
      const int qq = s & 3;
#pragma unroll
      for (int mt = 0; mt < 2; mt++) {
        v16bf af = load_frag16(&actb[mt * 16 + mrow][k2], hl);
#pragma unroll
        for (int nt = 0; nt < 2; nt++) {
          v16bf bf = load_frag16(&u.w2t[cur][wid * 32 + nt * 16 + mrow][0], hl);
          acc[mt][qq * 2 + nt] = wmma_bf16(af, bf, acc[mt][qq * 2 + nt]);
        }
      }
    }
    __syncthreads();  // region0 reads done before next chunk's G1 staging
  }

  // gate-scale and accumulate (each out element gets exactly 2 adds -> deterministic)
#pragma unroll
  for (int mt = 0; mt < 2; mt++)
#pragma unroll
    for (int j = 0; j < 8; j++) {
      int r = mt * 16 + hl * 8 + j;
      int tok = tok_s[r];
      float gv = gate_s[r];
      float* orow = out + (size_t)tok * HID;
#pragma unroll
      for (int q = 0; q < 8; q++) {
        int col = (q >> 1) * 256 + wid * 32 + (q & 1) * 16 + mrow;
        atomicAdd(orow + col, gv * acc[mt][q][j]);
      }
    }
}

// ---------------- host launcher ----------------
extern "C" void kernel_launch(void* const* d_in, const int* in_sizes, int n_in,
                              void* d_out, int out_size, void* d_ws, size_t ws_size,
                              hipStream_t stream) {
  const float* x  = (const float*)d_in[0];   // [8192,1024]
  const float* wr = (const float*)d_in[1];   // [1024,8]
  const float* w1 = (const float*)d_in[2];   // [8,1024,4096]
  const float* w2 = (const float*)d_in[3];   // [8,2048,1024]
  float* out = (float*)d_out;                // [8192,1024]

  char* ws = (char*)d_ws;
  size_t o = 0;
  int* counts = (int*)(ws + o);              o += 256;
  int* tok_list = (int*)(ws + o);            o += (size_t)NEXP * T_TOK * 4;
  float* gate_list = (float*)(ws + o);       o += (size_t)NEXP * T_TOK * 4;
  unsigned short* xb = (unsigned short*)(ws + o);   o += (size_t)T_TOK * HID * 2;
  unsigned short* w1s = (unsigned short*)(ws + o);  o += (size_t)NEXP * HID * W1COLS * 2;
  unsigned short* w2s = (unsigned short*)(ws + o);  // + NEXP*FFN*HID*2  (~118MB total)

  (void)in_sizes; (void)n_in; (void)out_size; (void)ws_size;

  moe_init_kernel<<<2048, 256, 0, stream>>>((float4*)out, (T_TOK * HID) / 4,
                                            counts);
  conv_w1_kernel<<<(NEXP * 32 * W1COLS) / 256, 256, 0, stream>>>(w1, w1s);
  conv_w2_kernel<<<(NEXP * 64 * HID) / 256, 256, 0, stream>>>(w2, w2s);
  moe_router_kernel<<<T_TOK / 8, 256, 0, stream>>>(x, wr, counts, tok_list,
                                                   gate_list, xb);
  moe_expert_kernel<<<NEXP * 256, 256, 0, stream>>>(xb, w1s, w2s, counts,
                                                    tok_list, gate_list, out);
}
